// NeuSRenderer_11587821764705
// MI455X (gfx1250) — matrix operand order repaired and checked by reading, not compiled, not run on hardware
//
#include <hip/hip_runtime.h>
#include <math.h>
#include <stdint.h>

typedef __attribute__((ext_vector_type(16))) _Float16 v16h;
typedef __attribute__((ext_vector_type(8)))  _Float16 v8h;
typedef __attribute__((ext_vector_type(8)))  float    v8f;
typedef __attribute__((ext_vector_type(4)))  unsigned int u32x4;
typedef __attribute__((ext_vector_type(8)))  int      i32x8;
typedef __attribute__((ext_vector_type(4)))  int      i32x4;

#define NUM_RAYS 4096
#define HID 128
#define SAMPLE_DISTF 0.03125f   /* 2/64 */

__device__ __forceinline__ float softplusf(float x) {
    return fmaxf(x, 0.f) + log1pf(expf(-fabsf(x)));
}
__device__ __forceinline__ float sigmoidf(float x) {
    return 1.f / (1.f + expf(-x));
}

// ---- Tensor Data Mover: DMA one 32KB f16 weight matrix (16384 elems) into LDS.
// D# group0: count=1 | lds_addr | global_addr(57b) | type=2
// D# group1: data_size=1(2B), tensor 16384x1, tile 16384x1, dim0_stride=16384
__device__ __forceinline__ void tdm_load_w(const _Float16* gsrc, _Float16* lds_dst) {
    unsigned long long ga = (unsigned long long)(uintptr_t)gsrc;
    unsigned int lds_off = (unsigned int)(uintptr_t)lds_dst;   // addr[31:0] = LDS offset
    u32x4 g0;
    g0[0] = 1u;                                            // count=1, user mode
    g0[1] = lds_off;                                       // lds_addr [63:32]
    g0[2] = (unsigned int)(ga & 0xFFFFFFFFu);              // global_addr [95:64]
    g0[3] = (unsigned int)((ga >> 32) & 0x1FFFFFFu)        // global_addr [120:96]
            | 0x80000000u;                                 // type=2 [127:126]
    i32x8 g1;
    g1[0] = 0x10000;                 // workgroup_mask=0, data_size=1 (2 bytes)
    g1[1] = (int)(16384u << 16);     // tensor_dim0[15:0] in [63:48]
    g1[2] = 0x10000;                 // tensor_dim0[31:16]=0, tensor_dim1[15:0]=1
    g1[3] = (int)(16384u << 16);     // tensor_dim1[31:16]=0, tile_dim0=16384
    g1[4] = 1;                       // tile_dim1=1, tile_dim2=0
    g1[5] = 16384;                   // tensor_dim0_stride[31:0]
    g1[6] = 0;
    g1[7] = 0;
    i32x4 gz = {0, 0, 0, 0};
#if defined(__clang_major__) && (__clang_major__ >= 23)
    i32x8 gz8 = {0, 0, 0, 0, 0, 0, 0, 0};
    __builtin_amdgcn_tensor_load_to_lds(g0, g1, gz, gz, gz8, 0);
#else
    __builtin_amdgcn_tensor_load_to_lds(g0, g1, gz, gz, 0);
#endif
}

// A fragment: 16x32 f16 (MxK), row m per lane, k = 16*(lane/16) + s (contiguous order).
__device__ __forceinline__ v16h load_a_frag(const _Float16* base, int m, int k0, int lh) {
    const _Float16* p = base + m * HID + k0 + lh * 16;
    v8h lo = *(const v8h*)p;
    v8h hi = *(const v8h*)(p + 8);
    v16h a;
#pragma unroll
    for (int i = 0; i < 8; i++) { a[i] = lo[i]; a[i + 8] = hi[i]; }
    return a;
}
// B fragment: 32x16 f16 (KxN) from column-major storage cm[n][k].
__device__ __forceinline__ v16h load_b_frag(const _Float16* cm, int n, int k0, int lh) {
    const _Float16* p = cm + n * HID + k0 + lh * 16;
    return *(const v16h*)p;
}

__global__ void k_prep_w2(const float* __restrict__ W2,
                          _Float16* __restrict__ w2cm,
                          _Float16* __restrict__ w2rm) {
    int i = blockIdx.x * blockDim.x + threadIdx.x;
    if (i < HID * HID) {
        int r = i / HID, c = i % HID;
        float v = W2[i];
        w2rm[i] = (_Float16)v;
        w2cm[c * HID + r] = (_Float16)v;
    }
}

__global__ void k_init_z(const float* __restrict__ nearp, const float* __restrict__ farp,
                         float* __restrict__ z) {
    int i = blockIdx.x * blockDim.x + threadIdx.x;
    if (i < NUM_RAYS * 64) {
        int ray = i >> 6, k = i & 63;
        float t = (float)k / 63.f;
        z[i] = nearp[ray] + (farp[ray] - nearp[ray]) * t;
    }
}

// Forward MLP: 128 points per block, 8 waves x 16 rows. pts = o + d*z.
__global__ void __launch_bounds__(256)
k_mlp_fwd(const float* __restrict__ ro, const float* __restrict__ rd,
          const float* __restrict__ zbuf, int S,
          const float* __restrict__ W1, const float* __restrict__ b1,
          const _Float16* __restrict__ w2cm,
          const float* __restrict__ b2, const float* __restrict__ W3,
          const float* __restrict__ b3,
          float* __restrict__ sdf_out) {
    __shared__ __align__(32) _Float16 sWB[HID * HID];
    __shared__ __align__(32) _Float16 sH1[128 * HID];
    __shared__ float sRed[8 * 32 * 8];
    __shared__ float sB2[HID], sW3[HID];

    int t = threadIdx.x;
    int p0 = blockIdx.x * 128;

    __builtin_prefetch(W1, 0, 3);
    if (t < 32) {   // wave 0 issues one TDM descriptor for the whole W2 tile
        tdm_load_w(w2cm, sWB);
        __builtin_amdgcn_s_wait_tensorcnt(0);
    }
    if (t < HID) { sB2[t] = b2[t]; sW3[t] = W3[t]; }

    {   // layer 1: each thread does 64 hidden units of one point
        int pl = t >> 1;
        int jb = (t & 1) * 64;
        int p = p0 + pl;
        unsigned ray = (unsigned)p / (unsigned)S;
        float zv = zbuf[p];
        float px = ro[ray * 3 + 0] + rd[ray * 3 + 0] * zv;
        float py = ro[ray * 3 + 1] + rd[ray * 3 + 1] * zv;
        float pz = ro[ray * 3 + 2] + rd[ray * 3 + 2] * zv;
        for (int j = 0; j < 64; j++) {
            int jj = jb + j;
            float z1 = px * W1[jj] + py * W1[HID + jj] + pz * W1[2 * HID + jj] + b1[jj];
            sH1[pl * HID + jj] = (_Float16)softplusf(z1);
        }
    }
    __syncthreads();

    int lane = t & 31, wave = t >> 5;
    int lh = lane >> 4;
    int m0 = wave * 16;
    int mrow = m0 + (lane & 15);

    float psdf[8];
#pragma unroll
    for (int r = 0; r < 8; r++) psdf[r] = 0.f;

    for (int nt = 0; nt < 8; nt++) {
        int n0 = nt * 16;
        v8f acc = {};
#pragma unroll
        for (int kt = 0; kt < 4; kt++) {
            v16h a = load_a_frag(sH1, mrow, kt * 32, lh);
            v16h b = load_b_frag(sWB, n0 + (lane & 15), kt * 32, lh);
            acc = __builtin_amdgcn_wmma_f32_16x16x32_f16(false, a, false, b,
                                                         (short)0, acc, false, false);
        }
        int col = n0 + (lane & 15);
        float bb = sB2[col], w3 = sW3[col];
#pragma unroll
        for (int r = 0; r < 8; r++) {
            float h2 = softplusf(acc[r] + bb);
            psdf[r] += h2 * w3;
        }
    }
#pragma unroll
    for (int r = 0; r < 8; r++) sRed[(wave * 32 + lane) * 8 + r] = psdf[r];
    __syncthreads();
    if (t < 128) {   // deterministic reduction over the 16 contributing lanes
        int w = t >> 4, rr = t & 15;
        int base = (rr < 8) ? 0 : 16, r = rr & 7;
        float s = b3[0];
        for (int l = 0; l < 16; l++) s += sRed[(w * 32 + base + l) * 8 + r];
        sdf_out[p0 + t] = s;
    }
}

// Fused forward + input-gradient MLP at midpoints (S=128).
__global__ void __launch_bounds__(256)
k_mlp_fwd_grad(const float* __restrict__ ro, const float* __restrict__ rd,
               const float* __restrict__ midz,
               const float* __restrict__ W1, const float* __restrict__ b1,
               const _Float16* __restrict__ w2cm, const _Float16* __restrict__ w2rm,
               const float* __restrict__ b2, const float* __restrict__ W3,
               const float* __restrict__ b3,
               float* __restrict__ sdf_out, float* __restrict__ grad_out) {
    __shared__ __align__(32) _Float16 sWB[HID * HID];
    __shared__ __align__(32) _Float16 sH1[128 * HID];
    __shared__ __align__(32) _Float16 sS1[128 * HID];
    __shared__ __align__(32) _Float16 sD2[128 * HID];
    __shared__ float sRed[8 * 32 * 24];
    __shared__ float sB2[HID], sW3c[HID], sW1c[3 * HID];

    int t = threadIdx.x;
    int p0 = blockIdx.x * 128;

    __builtin_prefetch(W1, 0, 3);
    if (t < 32) {
        tdm_load_w(w2cm, sWB);
        __builtin_amdgcn_s_wait_tensorcnt(0);
    }
    if (t < HID) { sB2[t] = b2[t]; sW3c[t] = W3[t]; }
    for (int i = t; i < 3 * HID; i += 256) sW1c[i] = W1[i];

    {   // layer 1 (keep softplus value and sigmoid derivative)
        int pl = t >> 1;
        int jb = (t & 1) * 64;
        int p = p0 + pl;
        unsigned ray = (unsigned)p / 128u;
        float zv = midz[p];
        float px = ro[ray * 3 + 0] + rd[ray * 3 + 0] * zv;
        float py = ro[ray * 3 + 1] + rd[ray * 3 + 1] * zv;
        float pz = ro[ray * 3 + 2] + rd[ray * 3 + 2] * zv;
        for (int j = 0; j < 64; j++) {
            int jj = jb + j;
            float z1 = px * W1[jj] + py * W1[HID + jj] + pz * W1[2 * HID + jj] + b1[jj];
            sH1[pl * HID + jj] = (_Float16)softplusf(z1);
            sS1[pl * HID + jj] = (_Float16)sigmoidf(z1);
        }
    }
    __syncthreads();

    int lane = t & 31, wave = t >> 5;
    int lh = lane >> 4;
    int m0 = wave * 16;
    int mrow = m0 + (lane & 15);

    float psdf[8];
#pragma unroll
    for (int r = 0; r < 8; r++) psdf[r] = 0.f;

    // GEMM1: Z2 = H1 @ W2 ; h2 = softplus, d2 = sigmoid(z2)*W3[col]
    for (int nt = 0; nt < 8; nt++) {
        int n0 = nt * 16;
        v8f acc = {};
#pragma unroll
        for (int kt = 0; kt < 4; kt++) {
            v16h a = load_a_frag(sH1, mrow, kt * 32, lh);
            v16h b = load_b_frag(sWB, n0 + (lane & 15), kt * 32, lh);
            acc = __builtin_amdgcn_wmma_f32_16x16x32_f16(false, a, false, b,
                                                         (short)0, acc, false, false);
        }
        int col = n0 + (lane & 15);
        float bb = sB2[col], w3 = sW3c[col];
#pragma unroll
        for (int r = 0; r < 8; r++) {
            float z2 = acc[r] + bb;
            psdf[r] += softplusf(z2) * w3;
            int m = m0 + r + lh * 8;
            sD2[m * HID + col] = (_Float16)(sigmoidf(z2) * w3);
        }
    }
#pragma unroll
    for (int r = 0; r < 8; r++) sRed[(wave * 32 + lane) * 8 + r] = psdf[r];
    __syncthreads();
    if (t < 128) {
        int w = t >> 4, rr = t & 15;
        int base = (rr < 8) ? 0 : 16, r = rr & 7;
        float s = b3[0];
        for (int l = 0; l < 16; l++) s += sRed[(w * 32 + base + l) * 8 + r];
        sdf_out[p0 + t] = s;
    }
    __syncthreads();   // all GEMM1 reads of sWB are done past this point

    if (t < 32) {      // DMA the W2^T tile (W2 row-major) over the weight buffer
        tdm_load_w(w2rm, sWB);
        __builtin_amdgcn_s_wait_tensorcnt(0);
    }
    __syncthreads();

    float gp[24];
#pragma unroll
    for (int i = 0; i < 24; i++) gp[i] = 0.f;

    // GEMM2: T = D2 @ W2^T ; d1 = T * sigmoid(z1) ; grad += d1 * W1^T
    for (int nt = 0; nt < 8; nt++) {
        int n0 = nt * 16;
        v8f acc = {};
#pragma unroll
        for (int kt = 0; kt < 4; kt++) {
            v16h a = load_a_frag(sD2, mrow, kt * 32, lh);
            v16h b = load_b_frag(sWB, n0 + (lane & 15), kt * 32, lh);
            acc = __builtin_amdgcn_wmma_f32_16x16x32_f16(false, a, false, b,
                                                         (short)0, acc, false, false);
        }
        int col = n0 + (lane & 15);
        float w1x = sW1c[col], w1y = sW1c[HID + col], w1z = sW1c[2 * HID + col];
#pragma unroll
        for (int r = 0; r < 8; r++) {
            int m = m0 + r + lh * 8;
            float d1 = acc[r] * (float)sS1[m * HID + col];
            gp[r * 3 + 0] += d1 * w1x;
            gp[r * 3 + 1] += d1 * w1y;
            gp[r * 3 + 2] += d1 * w1z;
        }
    }
#pragma unroll
    for (int i = 0; i < 24; i++) sRed[(wave * 32 + lane) * 24 + i] = gp[i];
    __syncthreads();
    if (t < 128) {
        int w = t >> 4, rr = t & 15;
        int base = (rr < 8) ? 0 : 16, r = rr & 7;
        float g0 = 0.f, g1 = 0.f, g2 = 0.f;
        for (int l = 0; l < 16; l++) {
            const float* q = &sRed[(w * 32 + base + l) * 24 + r * 3];
            g0 += q[0]; g1 += q[1]; g2 += q[2];
        }
        int p = p0 + t;
        grad_out[p * 3 + 0] = g0;
        grad_out[p * 3 + 1] = g1;
        grad_out[p * 3 + 2] = g2;
    }
}

// One thread per ray: NeuS upsample weights + inverse-CDF sampling of 16 new z.
__global__ void k_upsample(const float* __restrict__ ro, const float* __restrict__ rd,
                           const float* __restrict__ z, const float* __restrict__ sdf,
                           int S, float inv_s, float* __restrict__ new_z) {
    int ray = blockIdx.x * blockDim.x + threadIdx.x;
    if (ray >= NUM_RAYS) return;
    const float* zr = z + ray * S;
    const float* sr = sdf + ray * S;
    float ox = ro[ray * 3], oy = ro[ray * 3 + 1], oz = ro[ray * 3 + 2];
    float dx = rd[ray * 3], dy = rd[ray * 3 + 1], dz = rd[ray * 3 + 2];

    float w[127];
    float z0 = zr[0];
    float px = ox + dx * z0, py = oy + dy * z0, pz = oz + dz * z0;
    float r0 = sqrtf(px * px + py * py + pz * pz);
    float prev_cos = 0.f;
    float trans = 1.f;
    for (int j = 0; j < S - 1; j++) {
        float zz0 = zr[j], z1 = zr[j + 1];
        float qx = ox + dx * z1, qy = oy + dy * z1, qz = oz + dz * z1;
        float r1 = sqrtf(qx * qx + qy * qy + qz * qz);
        float inside = (r0 < 1.f || r1 < 1.f) ? 1.f : 0.f;
        float s0 = sr[j], s1 = sr[j + 1];
        float mid = (s0 + s1) * 0.5f;
        float cosr = (s1 - s0) / (z1 - zz0 + 1e-5f);
        float cv = fminf(prev_cos, cosr);
        prev_cos = cosr;
        cv = fminf(fmaxf(cv, -1000.f), 0.f) * inside;
        float dist = z1 - zz0;
        float e0 = sigmoidf((mid - cv * dist * 0.5f) * inv_s);
        float e1 = sigmoidf((mid + cv * dist * 0.5f) * inv_s);
        float alpha = (e0 - e1 + 1e-5f) / (e0 + 1e-5f);
        w[j] = alpha * trans;
        trans *= (1.f - alpha + 1e-7f);
        r0 = r1;
    }
    // sample_pdf
    float cdf[128];
    float tot = 0.f;
    for (int j = 0; j < S - 1; j++) { w[j] += 1e-5f; tot += w[j]; }
    cdf[0] = 0.f;
    for (int j = 0; j < S - 1; j++) cdf[j + 1] = cdf[j] + w[j] / tot;
    for (int k = 0; k < 16; k++) {
        float u = (1.f + 2.f * (float)k) / 32.f;
        int ind = 0;
        for (int j2 = 0; j2 < S; j2++) if (cdf[j2] <= u) ind = j2 + 1;
        int below = ind - 1; if (below < 0) below = 0;
        int above = ind;     if (above > S - 1) above = S - 1;
        float cb = cdf[below], ca = cdf[above];
        float bb = zr[below], ba = zr[above];
        float den = ca - cb; if (den < 1e-5f) den = 1.f;
        float tt = (u - cb) / den;
        new_z[ray * 16 + k] = bb + tt * (ba - bb);
    }
}

// Stable merge of sorted old samples with sorted new samples (+carried sdf).
__global__ void k_merge(const float* __restrict__ z, const float* __restrict__ sdf,
                        const float* __restrict__ nz, const float* __restrict__ nsdf,
                        int S, int last,
                        float* __restrict__ z_out, float* __restrict__ sdf_out) {
    int ray = blockIdx.x * blockDim.x + threadIdx.x;
    if (ray >= NUM_RAYS) return;
    const float* za = z + ray * S;
    const float* sa = sdf + ray * S;
    const float* zb = nz + ray * 16;
    const float* sb = nsdf + ray * 16;
    float* zo = z_out + ray * (S + 16);
    float* so = sdf_out + ray * (S + 16);
    int i = 0, j = 0;
    for (int k = 0; k < S + 16; k++) {
        bool takeA = (j >= 16) || (i < S && za[i] <= zb[j]);
        if (takeA) { zo[k] = za[i]; if (!last) so[k] = sa[i]; i++; }
        else       { zo[k] = zb[j]; if (!last) so[k] = sb[j]; j++; }
    }
}

__global__ void k_midz(const float* __restrict__ z, float* __restrict__ midz,
                       float* __restrict__ dists) {
    int i = blockIdx.x * blockDim.x + threadIdx.x;
    if (i >= NUM_RAYS * 128) return;
    int s = i & 127;
    float d = (s < 127) ? (z[i + 1] - z[i]) : SAMPLE_DISTF;
    dists[i] = d;
    midz[i] = z[i] + 0.5f * d;
}

// One thread per ray: transmittance scan -> depth.
__global__ void k_render(const float* __restrict__ z, const float* __restrict__ dists,
                         const float* __restrict__ sdfm, const float* __restrict__ grad,
                         const float* __restrict__ rd, const float* __restrict__ varp,
                         float* __restrict__ depth_out) {
    int ray = blockIdx.x * blockDim.x + threadIdx.x;
    if (ray >= NUM_RAYS) return;
    float inv_s = expf(10.f * varp[0]);
    inv_s = fminf(fmaxf(inv_s, 1e-6f), 1e6f);
    float dx = rd[ray * 3], dy = rd[ray * 3 + 1], dz = rd[ray * 3 + 2];
    float trans = 1.f, depth = 0.f;
    for (int s = 0; s < 128; s++) {
        int p = ray * 128 + s;
        float g0 = grad[p * 3], g1 = grad[p * 3 + 1], g2 = grad[p * 3 + 2];
        float tc = dx * g0 + dy * g1 + dz * g2;
        float ic = -fmaxf(-tc * 0.5f + 0.5f, 0.f);   // COS_ANNEAL_RATIO = 0
        float sd = sdfm[p];
        float dd = dists[p];
        float e0 = sigmoidf((sd - ic * dd * 0.5f) * inv_s);
        float e1 = sigmoidf((sd + ic * dd * 0.5f) * inv_s);
        float alpha = (e0 - e1 + 1e-5f) / (e0 + 1e-5f);
        alpha = fminf(fmaxf(alpha, 0.f), 1.f);
        float wgt = alpha * trans;
        depth += wgt * z[p];
        trans *= (1.f - alpha + 1e-7f);
    }
    depth_out[ray] = depth;
}

extern "C" void kernel_launch(void* const* d_in, const int* in_sizes, int n_in,
                              void* d_out, int out_size, void* d_ws, size_t ws_size,
                              hipStream_t stream) {
    const float* ro    = (const float*)d_in[0];
    const float* rd    = (const float*)d_in[1];
    const float* nearp = (const float*)d_in[2];
    const float* farp  = (const float*)d_in[3];
    const float* W1    = (const float*)d_in[4];
    const float* b1    = (const float*)d_in[5];
    const float* W2    = (const float*)d_in[6];
    const float* b2    = (const float*)d_in[7];
    const float* W3    = (const float*)d_in[8];
    const float* b3    = (const float*)d_in[9];
    const float* var   = (const float*)d_in[10];

    float* f = (float*)d_ws;
    size_t off = 0;
    float* z_a     = f + off; off += (size_t)NUM_RAYS * 128;
    float* z_b     = f + off; off += (size_t)NUM_RAYS * 128;
    float* sdf_a   = f + off; off += (size_t)NUM_RAYS * 128;
    float* sdf_b   = f + off; off += (size_t)NUM_RAYS * 128;
    float* new_z   = f + off; off += (size_t)NUM_RAYS * 16;
    float* new_sdf = f + off; off += (size_t)NUM_RAYS * 16;
    float* mid_z   = f + off; off += (size_t)NUM_RAYS * 128;
    float* dists   = f + off; off += (size_t)NUM_RAYS * 128;
    float* sdf_mid = f + off; off += (size_t)NUM_RAYS * 128;
    _Float16* w2cm = (_Float16*)(f + off); off += (HID * HID) / 2;
    _Float16* w2rm = (_Float16*)(f + off); off += (HID * HID) / 2;

    float* depths = (float*)d_out;
    float* grads  = (float*)d_out + NUM_RAYS;

    k_prep_w2<<<(HID * HID + 255) / 256, 256, 0, stream>>>(W2, w2cm, w2rm);
    k_init_z<<<(NUM_RAYS * 64 + 255) / 256, 256, 0, stream>>>(nearp, farp, z_a);
    k_mlp_fwd<<<(NUM_RAYS * 64) / 128, 256, 0, stream>>>(ro, rd, z_a, 64,
                                                         W1, b1, w2cm, b2, W3, b3, sdf_a);
    float* zc = z_a; float* zn = z_b;
    float* sc = sdf_a; float* sn = sdf_b;
    int S = 64;
    for (int i = 0; i < 4; i++) {
        float inv_s = 64.f * (float)(1 << i);
        k_upsample<<<(NUM_RAYS + 63) / 64, 64, 0, stream>>>(ro, rd, zc, sc, S, inv_s, new_z);
        int last = (i == 3);
        if (!last) {
            k_mlp_fwd<<<(NUM_RAYS * 16) / 128, 256, 0, stream>>>(ro, rd, new_z, 16,
                                                                 W1, b1, w2cm, b2, W3, b3, new_sdf);
        }
        k_merge<<<(NUM_RAYS + 63) / 64, 64, 0, stream>>>(zc, sc, new_z, new_sdf, S, last, zn, sn);
        float* tz = zc; zc = zn; zn = tz;
        float* ts = sc; sc = sn; sn = ts;
        S += 16;
    }
    // zc now holds 128 sorted samples per ray
    k_midz<<<(NUM_RAYS * 128 + 255) / 256, 256, 0, stream>>>(zc, mid_z, dists);
    k_mlp_fwd_grad<<<(NUM_RAYS * 128) / 128, 256, 0, stream>>>(ro, rd, mid_z,
                                                               W1, b1, w2cm, w2rm,
                                                               b2, W3, b3, sdf_mid, grads);
    k_render<<<(NUM_RAYS + 63) / 64, 64, 0, stream>>>(zc, dists, sdf_mid, grads, rd, var, depths);
}
